// CrossAttnBlock_64355789963922
// MI455X (gfx1250) — matrix operand order repaired and checked by reading, not compiled, not run on hardware
//
#include <hip/hip_runtime.h>

typedef __attribute__((ext_vector_type(16))) __bf16 v16bf;
typedef __attribute__((ext_vector_type(8)))  __bf16 v8bf;
typedef __attribute__((ext_vector_type(8)))  float  v8f;
typedef __attribute__((ext_vector_type(4)))  float  v4f;
typedef unsigned short u16;
typedef __attribute__((ext_vector_type(8)))  u16 v8u;
typedef __attribute__((ext_vector_type(16))) u16 v16u;

constexpr int B_   = 16;
constexpr int LQ_  = 4096;
constexpr int D_   = 512;
constexpr int CD_  = 768;
constexpr int H_   = 8;
constexpr int HD_  = 64;
constexpr int LK_  = 77;
constexpr int LKP_ = 96;      // padded to multiple of 32 (WMMA K)
constexpr int MT_  = 32;      // query rows per workgroup
constexpr int QZ_STRIDE = 528;  // 512 + pad, 16B-aligned rows
constexpr int P_STRIDE  = 104;  // 96 + pad, 16B-aligned rows

// native fptrunc -> v_cvt_*bf16* hardware converts (RNE), no manual bit math
__device__ __forceinline__ u16 f2bf(float f) {
  __bf16 h = (__bf16)f;
  return __builtin_bit_cast(u16, h);
}

// A/B fragment from two contiguous 16-byte bf16 chunks (LDS or global)
__device__ __forceinline__ v16bf frag_u16(const u16* p0, const u16* p1) {
  v8u lo = *(const v8u*)p0;
  v8u hi = *(const v8u*)p1;
  v16u c;
#pragma unroll
  for (int i = 0; i < 8; ++i) { c[i] = lo[i]; c[i + 8] = hi[i]; }
  return __builtin_bit_cast(v16bf, c);
}

// A fragment built from fp32 global memory (two 8-float chunks) via HW converts
__device__ __forceinline__ v16bf frag_f32(const float* p0, const float* p1) {
  v4f a0 = *(const v4f*)p0, a1 = *(const v4f*)(p0 + 4);
  v4f b0 = *(const v4f*)p1, b1 = *(const v4f*)(p1 + 4);
  v16bf c;
#pragma unroll
  for (int i = 0; i < 4; ++i) {
    c[i]      = (__bf16)a0[i];
    c[4 + i]  = (__bf16)a1[i];
    c[8 + i]  = (__bf16)b0[i];
    c[12 + i] = (__bf16)b1[i];
  }
  return c;
}

__device__ __forceinline__ v8f wmma_bf(v16bf a, v16bf b, v8f c) {
  return __builtin_amdgcn_wmma_f32_16x16x32_bf16(false, a, false, b, (short)0, c,
                                                 false, false);
}

// ---------------- prep kernels ----------------

__global__ void cvt_bf16(const float* __restrict__ src, u16* __restrict__ dst, int n) {
  int i = blockIdx.x * blockDim.x + threadIdx.x;
  if (i < n) dst[i] = f2bf(src[i]);
}

// K[b][h][jp][d] (row-major d) and Vt[b][h][d][jp] (row-major jp), zero-padded jp>=77
__global__ void prep_kv(const float* __restrict__ cond, const float* __restrict__ wk,
                        const float* __restrict__ wv, u16* __restrict__ kmat,
                        u16* __restrict__ vt) {
  int t  = blockIdx.x * blockDim.x + threadIdx.x;  // B*LKP*D
  int o  = t % D_;
  int rem = t / D_;
  int jp = rem % LKP_;
  int b  = rem / LKP_;
  if (b >= B_) return;
  float kv = 0.f, vv = 0.f;
  if (jp < LK_) {
    const float* c  = cond + ((size_t)b * LK_ + jp) * CD_;
    const float* pk = wk + (size_t)o * CD_;
    const float* pv = wv + (size_t)o * CD_;
    for (int i = 0; i < CD_; i += 4) {
      v4f cv = *(const v4f*)(c + i);
      v4f kw = *(const v4f*)(pk + i);
      v4f vw = *(const v4f*)(pv + i);
#pragma unroll
      for (int q = 0; q < 4; ++q) { kv += cv[q] * kw[q]; vv += cv[q] * vw[q]; }
    }
  }
  int h = o / HD_, dd = o % HD_;
  kmat[(((size_t)b * H_ + h) * LKP_ + jp) * HD_ + dd] = f2bf(kv);
  vt[(((size_t)b * H_ + h) * HD_ + dd) * LKP_ + jp]   = f2bf(vv);
}

// ---------------- fused Q-proj -> attention -> out-proj ----------------

__global__ __launch_bounds__(128) void fused_xattn(
    const float* __restrict__ x, const u16* __restrict__ wq,
    const u16* __restrict__ kmat, const u16* __restrict__ vt,
    const u16* __restrict__ wo, float* __restrict__ out) {
  __shared__ __align__(16) u16 QZ[MT_ * QZ_STRIDE];       // Q, later aliased by Z
  __shared__ __align__(16) u16 P[4][MT_ * P_STRIDE];      // per-wave softmax probs

  const int b   = blockIdx.x / (LQ_ / MT_);
  const int r0  = (blockIdx.x % (LQ_ / MT_)) * MT_;
  const int tid = threadIdx.x;
  const int wave = tid >> 5, lane = tid & 31;
  const int hl = lane >> 4, ln = lane & 15;   // half-group, lane-in-16
  const int mtile = wave & 1;                 // 0..1 (rows mtile*16..+15)
  const int ngrp  = wave >> 1;                // 0..1 (16 ntiles each)

  // ---- Phase 1: Q = X @ Wq^T ----
  {
    v8f acc[16] = {};
    const float* xrow = x + ((size_t)b * LQ_ + (r0 + mtile * 16 + ln)) * D_;
#pragma unroll 1
    for (int ks = 0; ks < 16; ++ks) {
      const float* pa = xrow + ks * 32 + hl * 8;
      v16bf a = frag_f32(pa, pa + 16);
#pragma unroll
      for (int nt = 0; nt < 16; ++nt) {
        const int o = (ngrp * 16 + nt) * 16 + ln;
        const u16* pb = wq + (size_t)o * D_ + ks * 32 + hl * 8;
        v16bf bb = frag_u16(pb, pb + 16);
        acc[nt] = wmma_bf(a, bb, acc[nt]);
      }
    }
#pragma unroll
    for (int nt = 0; nt < 16; ++nt) {
      const int col = (ngrp * 16 + nt) * 16 + ln;
#pragma unroll
      for (int r = 0; r < 8; ++r) {
        const int row = mtile * 16 + hl * 8 + r;
        QZ[row * QZ_STRIDE + col] = f2bf(acc[nt][r]);
      }
    }
  }
  __syncthreads();

  // ---- Phase 2: attention; wave w handles heads 2w, 2w+1 ----
  for (int hh = 0; hh < 2; ++hh) {
    const int h = wave * 2 + hh;
    v8f s[2][6] = {};
    // scores = Qh @ Kh^T
#pragma unroll
    for (int ks = 0; ks < 2; ++ks) {
#pragma unroll
      for (int mt = 0; mt < 2; ++mt) {
        const u16* pa = &QZ[(mt * 16 + ln) * QZ_STRIDE + h * HD_ + ks * 32 + hl * 8];
        v16bf a = frag_u16(pa, pa + 16);
#pragma unroll
        for (int nt = 0; nt < 6; ++nt) {
          const int j = nt * 16 + ln;
          const u16* pb =
              kmat + (((size_t)b * H_ + h) * LKP_ + j) * HD_ + ks * 32 + hl * 8;
          v16bf bb = frag_u16(pb, pb + 16);
          s[mt][nt] = wmma_bf(a, bb, s[mt][nt]);
        }
      }
    }
    // row softmax (scale 1/sqrt(64), mask j>=77)
    const float scale = 0.125f;
#pragma unroll
    for (int mt = 0; mt < 2; ++mt) {
#pragma unroll
      for (int r = 0; r < 8; ++r) {
        float v[6];
        float vmax = -__builtin_inff();
#pragma unroll
        for (int nt = 0; nt < 6; ++nt) {
          const int j = nt * 16 + ln;
          float t = s[mt][nt][r] * scale;
          t = (j < LK_) ? t : -__builtin_inff();
          v[nt] = t;
          vmax = fmaxf(vmax, t);
        }
#pragma unroll
        for (int m = 1; m < 16; m <<= 1) vmax = fmaxf(vmax, __shfl_xor(vmax, m, 32));
        float sum = 0.f;
#pragma unroll
        for (int nt = 0; nt < 6; ++nt) {
          float p = __expf(v[nt] - vmax);
          v[nt] = p;
          sum += p;
        }
#pragma unroll
        for (int m = 1; m < 16; m <<= 1) sum += __shfl_xor(sum, m, 32);
        const float inv = 1.0f / sum;
        const int row = mt * 16 + hl * 8 + r;
#pragma unroll
        for (int nt = 0; nt < 6; ++nt)
          P[wave][row * P_STRIDE + nt * 16 + ln] = f2bf(v[nt] * inv);
      }
    }
    // out_h = P @ V  -> Z columns h*64.. (aliases QZ; same wave, in-order DS)
    v8f oacc[2][4] = {};
#pragma unroll
    for (int ks = 0; ks < 3; ++ks) {
#pragma unroll
      for (int mt = 0; mt < 2; ++mt) {
        const u16* pa = &P[wave][(mt * 16 + ln) * P_STRIDE + ks * 32 + hl * 8];
        v16bf a = frag_u16(pa, pa + 16);
#pragma unroll
        for (int nt = 0; nt < 4; ++nt) {
          const int dd = nt * 16 + ln;
          const u16* pb =
              vt + (((size_t)b * H_ + h) * HD_ + dd) * LKP_ + ks * 32 + hl * 8;
          v16bf bb = frag_u16(pb, pb + 16);
          oacc[mt][nt] = wmma_bf(a, bb, oacc[mt][nt]);
        }
      }
    }
#pragma unroll
    for (int mt = 0; mt < 2; ++mt)
#pragma unroll
      for (int nt = 0; nt < 4; ++nt) {
        const int col = h * HD_ + nt * 16 + ln;
#pragma unroll
        for (int r = 0; r < 8; ++r) {
          const int row = mt * 16 + hl * 8 + r;
          QZ[row * QZ_STRIDE + col] = f2bf(oacc[mt][nt][r]);
        }
      }
  }
  __syncthreads();

  // ---- Phase 3: out = Z @ Wo^T ----
  {
    v8f acc[16] = {};
#pragma unroll 1
    for (int ks = 0; ks < 16; ++ks) {
      const u16* pa = &QZ[(mtile * 16 + ln) * QZ_STRIDE + ks * 32 + hl * 8];
      v16bf a = frag_u16(pa, pa + 16);
#pragma unroll
      for (int nt = 0; nt < 16; ++nt) {
        const int o = (ngrp * 16 + nt) * 16 + ln;
        const u16* pb = wo + (size_t)o * D_ + ks * 32 + hl * 8;
        v16bf bb = frag_u16(pb, pb + 16);
        acc[nt] = wmma_bf(a, bb, acc[nt]);
      }
    }
#pragma unroll
    for (int nt = 0; nt < 16; ++nt) {
      const int col = (ngrp * 16 + nt) * 16 + ln;
#pragma unroll
      for (int r = 0; r < 8; ++r) {
        const int row = r0 + mtile * 16 + hl * 8 + r;
        out[((size_t)b * LQ_ + row) * D_ + col] = acc[nt][r];
      }
    }
  }
}

// ---------------- host launch ----------------

extern "C" void kernel_launch(void* const* d_in, const int* in_sizes, int n_in,
                              void* d_out, int out_size, void* d_ws, size_t ws_size,
                              hipStream_t stream) {
  (void)in_sizes; (void)n_in; (void)out_size; (void)ws_size;
  const float* x    = (const float*)d_in[0];
  const float* cond = (const float*)d_in[1];
  const float* w_q  = (const float*)d_in[2];
  const float* w_k  = (const float*)d_in[3];
  const float* w_v  = (const float*)d_in[4];
  const float* w_o  = (const float*)d_in[5];
  float* out = (float*)d_out;

  u16* wq_bf = (u16*)d_ws;                       // 512*512
  u16* wo_bf = wq_bf + (size_t)D_ * D_;          // 512*512
  u16* k_bf  = wo_bf + (size_t)D_ * D_;          // 16*8*96*64
  u16* vt_bf = k_bf + (size_t)B_ * H_ * LKP_ * HD_;

  cvt_bf16<<<(D_ * D_ + 255) / 256, 256, 0, stream>>>(w_q, wq_bf, D_ * D_);
  cvt_bf16<<<(D_ * D_ + 255) / 256, 256, 0, stream>>>(w_o, wo_bf, D_ * D_);
  prep_kv<<<(B_ * LKP_ * D_ + 255) / 256, 256, 0, stream>>>(cond, w_k, w_v, k_bf,
                                                            vt_bf);
  fused_xattn<<<B_ * (LQ_ / MT_), 128, 0, stream>>>(x, wq_bf, k_bf, vt_bf, wo_bf,
                                                    out);
}
